// TGCN_LSTM_31722628448348
// MI455X (gfx1250) — matrix operand order repaired and checked by reading, not compiled, not run on hardware
//
#include <hip/hip_runtime.h>
#include <hip/hip_bf16.h>

#define NN    30000
#define EE    480000
#define DIN   512
#define DOUT  512
#define NCOLS 1536          // 3 gates (i,g,o) * 512
#define LDA   520           // padded LDS row stride (bf16 elems): 520*2=1040B -> conflict-free

typedef __attribute__((ext_vector_type(16))) __bf16       bf16x16;
typedef __attribute__((ext_vector_type(8)))  float        f32x8;
typedef __attribute__((ext_vector_type(4)))  unsigned int u32x4;

union Frag { bf16x16 v; u32x4 u[2]; };

__device__ __forceinline__ float fsigmoid(float x) {
    return 1.0f / (1.0f + __expf(-x));
}
__device__ __forceinline__ float ftanh(float x) {
    float xc = fminf(fmaxf(x, -15.0f), 15.0f);
    float e  = __expf(2.0f * xc);
    return (e - 1.0f) / (e + 1.0f);
}

// ---------------- init: AGG = bc (broadcast per column), deg = 1.0 (self loop) -------------
__global__ void k_init(const float* __restrict__ bc_i, const float* __restrict__ bc_g,
                       const float* __restrict__ bc_o, float* __restrict__ AGG,
                       float* __restrict__ deg) {
    size_t idx = (size_t)blockIdx.x * blockDim.x + threadIdx.x;
    size_t total = (size_t)NN * NCOLS;
    if (idx < total) {
        unsigned c = (unsigned)(idx % NCOLS);
        float b = (c < 512u) ? bc_i[c] : (c < 1024u ? bc_g[c - 512u] : bc_o[c - 1024u]);
        AGG[idx] = b;
    }
    if (idx < NN) deg[idx] = 1.0f;
}

// ---------------- degree accumulation over real edges --------------------------------------
__global__ void k_deg(const long long* __restrict__ ei, const float* __restrict__ ew,
                      float* __restrict__ deg) {
    int e = blockIdx.x * blockDim.x + threadIdx.x;
    if (e >= EE) return;
    long long d = ei[EE + e];
    unsafeAtomicAdd(&deg[d], ew[e]);
}

// ---------------- deg -> d^{-1/2} in place (deg >= 1 always) -------------------------------
__global__ void k_dinv(float* __restrict__ deg) {
    int i = blockIdx.x * blockDim.x + threadIdx.x;
    if (i < NN) deg[i] = rsqrtf(deg[i]);
}

// ---------------- weight prep: transpose to [N][K] bf16 ------------------------------------
// Wct[n][k] = Wc_gate[k][n%512], n = gate*512 + nl   (gates i,g,o)
// Wlt[g][n][k] = Wl_gate[k][n]  for k < 512 (top half of the 1024x512 matrix; H==0)
__global__ void k_prep(const float* __restrict__ Wc_i, const float* __restrict__ Wc_g,
                       const float* __restrict__ Wc_o, const float* __restrict__ Wl_i,
                       const float* __restrict__ Wl_g, const float* __restrict__ Wl_o,
                       __bf16* __restrict__ Wct, __bf16* __restrict__ Wlt) {
    int idx = blockIdx.x * blockDim.x + threadIdx.x;
    const int WCT = NCOLS * DIN;
    if (idx < WCT) {
        int n = idx / DIN, k = idx % DIN;
        int g = n >> 9, nl = n & 511;
        const float* W = (g == 0) ? Wc_i : (g == 1 ? Wc_g : Wc_o);
        Wct[idx] = (__bf16)W[(size_t)k * DOUT + nl];
    } else if (idx < 2 * WCT) {
        int j = idx - WCT;
        int g = j / (DOUT * DOUT);
        int r = j % (DOUT * DOUT);
        int n = r / DOUT, k = r % DOUT;
        const float* W = (g == 0) ? Wl_i : (g == 1 ? Wl_g : Wl_o);
        Wlt[j] = (__bf16)W[(size_t)k * DOUT + n];
    }
}

// ---------------- GEMM1: XW(bf16)[30000x1536] = x @ [Wc_i|Wc_g|Wc_o] ----------------------
// grid (1875, 3); block 256 = 8 waves; block tile 16 x 512; wave tile 16 x 64 (4 wmma accums)
__global__ __launch_bounds__(256) void k_gemm1(const float* __restrict__ x,
                                               const __bf16* __restrict__ Wct,
                                               __bf16* __restrict__ XW) {
    __shared__ __bf16 ldsA[16 * LDA];
    const int tid   = threadIdx.x;
    const int mbase = blockIdx.x * 16;
    const int nchunk = blockIdx.y;                       // 0..2 (gate)
    // stage 16x512 A strip, f32 -> bf16
    #pragma unroll
    for (int i = 0; i < 32; ++i) {
        int idx = i * 256 + tid;                         // 0..8191
        int r = idx >> 9, c = idx & 511;
        ldsA[r * LDA + c] = (__bf16)x[(size_t)(mbase + r) * DIN + c];
    }
    __syncthreads();

    const int wave = tid >> 5, lane = tid & 31;
    const int half = lane >> 4, lanem = lane & 15;
    const int nbase = nchunk * 512 + wave * 64;

    f32x8 acc0 = {}, acc1 = {}, acc2 = {}, acc3 = {};
    const __bf16* arow = &ldsA[lanem * LDA + half * 8];
    #pragma unroll 4
    for (int k0 = 0; k0 < DIN; k0 += 32) {
        Frag a;
        a.u[0] = *(const u32x4*)(arow + k0);             // K = k0+{0..7}/{8..15}
        a.u[1] = *(const u32x4*)(arow + k0 + 16);        // K = k0+{16..23}/{24..31}
        const __bf16* bp = &Wct[(size_t)(nbase + lanem) * DIN + k0 + half * 8];
        Frag b0, b1, b2, b3;
        b0.u[0] = *(const u32x4*)(bp);              b0.u[1] = *(const u32x4*)(bp + 16);
        bp += 16 * DIN;
        b1.u[0] = *(const u32x4*)(bp);              b1.u[1] = *(const u32x4*)(bp + 16);
        bp += 16 * DIN;
        b2.u[0] = *(const u32x4*)(bp);              b2.u[1] = *(const u32x4*)(bp + 16);
        bp += 16 * DIN;
        b3.u[0] = *(const u32x4*)(bp);              b3.u[1] = *(const u32x4*)(bp + 16);
        acc0 = __builtin_amdgcn_wmma_f32_16x16x32_bf16(false, a.v, false, b0.v, (short)0, acc0, false, false);
        acc1 = __builtin_amdgcn_wmma_f32_16x16x32_bf16(false, a.v, false, b1.v, (short)0, acc1, false, false);
        acc2 = __builtin_amdgcn_wmma_f32_16x16x32_bf16(false, a.v, false, b2.v, (short)0, acc2, false, false);
        acc3 = __builtin_amdgcn_wmma_f32_16x16x32_bf16(false, a.v, false, b3.v, (short)0, acc3, false, false);
    }
    #pragma unroll
    for (int r = 0; r < 8; ++r) {
        int row = mbase + r + half * 8;
        size_t base = (size_t)row * NCOLS + nbase + lanem;
        XW[base]      = (__bf16)acc0[r];
        XW[base + 16] = (__bf16)acc1[r];
        XW[base + 32] = (__bf16)acc2[r];
        XW[base + 48] = (__bf16)acc3[r];
    }
}

// ---------------- sparse aggregation: one wave per (edge | self-loop) ----------------------
__global__ __launch_bounds__(256) void k_agg(const long long* __restrict__ ei,
                                             const float* __restrict__ ew,
                                             const float* __restrict__ dinv,
                                             const unsigned* __restrict__ XWp,
                                             float* __restrict__ AGG) {
    int w = (blockIdx.x * blockDim.x + threadIdx.x) >> 5;
    int lane = threadIdx.x & 31;
    if (w >= EE + NN) return;
    long long s, d; float wt;
    if (w < EE) { s = ei[w]; d = ei[EE + w]; wt = ew[w]; }
    else        { s = w - EE; d = s;         wt = 1.0f;  }
    float nrm = dinv[s] * wt * dinv[d];
    const unsigned* srow = XWp + (size_t)s * (NCOLS / 2);  // 2 bf16 per dword
    float* drow = AGG + (size_t)d * NCOLS;
    for (int c = lane; c < NCOLS / 2; c += 32) {
        unsigned p = srow[c];
        float lo = __uint_as_float(p << 16);               // bf16 -> f32 via shift
        float hi = __uint_as_float(p & 0xffff0000u);
        unsafeAtomicAdd(drow + 2 * c,     nrm * lo);
        unsafeAtomicAdd(drow + 2 * c + 1, nrm * hi);
    }
}

// ---------------- GEMM2 + gate fusion: out = {O, Hn, Cn} ----------------------------------
// grid 1875; per block: 16 rows, all 512 cols, 3 gates sequentially sharing LDS A strip
__global__ __launch_bounds__(256) void k_gemm2(const float* __restrict__ AGG,
                                               const __bf16* __restrict__ Wlt,
                                               const float* __restrict__ bl_i,
                                               const float* __restrict__ bl_g,
                                               const float* __restrict__ bl_o,
                                               float* __restrict__ out) {
    __shared__ __bf16 ldsA[16 * LDA];
    const int tid   = threadIdx.x;
    const int mbase = blockIdx.x * 16;
    const int wave = tid >> 5, lane = tid & 31;
    const int half = lane >> 4, lanem = lane & 15;
    const int nbase = wave * 64;

    f32x8 acc[3][4] = {};
    #pragma unroll
    for (int g = 0; g < 3; ++g) {
        if (g) __syncthreads();                     // previous gate done reading LDS
        #pragma unroll
        for (int i = 0; i < 32; ++i) {
            int idx = i * 256 + tid;
            int r = idx >> 9, c = idx & 511;
            ldsA[r * LDA + c] = (__bf16)AGG[(size_t)(mbase + r) * NCOLS + g * 512 + c];
        }
        __syncthreads();
        const __bf16* arow  = &ldsA[lanem * LDA + half * 8];
        const __bf16* wbase = Wlt + (size_t)g * DOUT * DOUT;
        #pragma unroll 4
        for (int k0 = 0; k0 < DOUT; k0 += 32) {
            Frag a;
            a.u[0] = *(const u32x4*)(arow + k0);
            a.u[1] = *(const u32x4*)(arow + k0 + 16);
            const __bf16* bp = wbase + (size_t)(nbase + lanem) * DOUT + k0 + half * 8;
            #pragma unroll
            for (int t = 0; t < 4; ++t) {
                Frag b;
                b.u[0] = *(const u32x4*)(bp);
                b.u[1] = *(const u32x4*)(bp + 16);
                acc[g][t] = __builtin_amdgcn_wmma_f32_16x16x32_bf16(
                    false, a.v, false, b.v, (short)0, acc[g][t], false, false);
                bp += (size_t)16 * DOUT;
            }
        }
    }
    // fused LSTM-gate epilogue (C=0, H=0): I=sig, G=tanh, O=sig; Cn=I*G; Hn=O*tanh(Cn)
    const size_t NS = (size_t)NN * DOUT;
    #pragma unroll
    for (int t = 0; t < 4; ++t) {
        int col = nbase + t * 16 + lanem;
        float bi = bl_i[col], bg = bl_g[col], bo = bl_o[col];
        #pragma unroll
        for (int r = 0; r < 8; ++r) {
            int row = mbase + r + half * 8;
            size_t o = (size_t)row * DOUT + col;
            float I  = fsigmoid(acc[0][t][r] + bi);
            float G  = ftanh(acc[1][t][r] + bg);
            float O  = fsigmoid(acc[2][t][r] + bo);
            float Cn = I * G;
            float Hn = O * ftanh(Cn);
            out[o]          = O;
            out[NS + o]     = Hn;
            out[2 * NS + o] = Cn;
        }
    }
}

extern "C" void kernel_launch(void* const* d_in, const int* in_sizes, int n_in,
                              void* d_out, int out_size, void* d_ws, size_t ws_size,
                              hipStream_t stream) {
    const float*     x    = (const float*)d_in[0];
    const long long* ei   = (const long long*)d_in[1];   // int64, [2][E]
    const float*     ew   = (const float*)d_in[2];
    const float*     Wc_i = (const float*)d_in[3];
    const float*     bc_i = (const float*)d_in[4];
    const float*     Wl_i = (const float*)d_in[5];
    const float*     bl_i = (const float*)d_in[6];
    // f-gate inputs (7..10) provably unused: C==0 and F not returned
    const float*     Wc_g = (const float*)d_in[11];
    const float*     bc_g = (const float*)d_in[12];
    const float*     Wl_g = (const float*)d_in[13];
    const float*     bl_g = (const float*)d_in[14];
    const float*     Wc_o = (const float*)d_in[15];
    const float*     bc_o = (const float*)d_in[16];
    const float*     Wl_o = (const float*)d_in[17];
    const float*     bl_o = (const float*)d_in[18];

    char* wsp = (char*)d_ws;
    auto alloc = [&](size_t bytes) {
        char* p = wsp; wsp += (bytes + 255) & ~(size_t)255; return p;
    };
    float*  deg = (float*)alloc((size_t)NN * 4);                 // deg -> dinv in place
    float*  AGG = (float*)alloc((size_t)NN * NCOLS * 4);         // 184 MB
    __bf16* XW  = (__bf16*)alloc((size_t)NN * NCOLS * 2);        // 92 MB (L2-resident)
    __bf16* Wct = (__bf16*)alloc((size_t)NCOLS * DIN * 2);
    __bf16* Wlt = (__bf16*)alloc((size_t)3 * DOUT * DOUT * 2);
    float*  out = (float*)d_out;

    {
        size_t total = (size_t)NN * NCOLS;
        k_init<<<(unsigned)((total + 255) / 256), 256, 0, stream>>>(bc_i, bc_g, bc_o, AGG, deg);
    }
    k_deg<<<(EE + 255) / 256, 256, 0, stream>>>(ei, ew, deg);
    k_dinv<<<(NN + 255) / 256, 256, 0, stream>>>(deg);
    k_prep<<<(2 * NCOLS * DIN + 255) / 256, 256, 0, stream>>>(Wc_i, Wc_g, Wc_o,
                                                              Wl_i, Wl_g, Wl_o, Wct, Wlt);
    dim3 g1(NN / 16, 3);
    k_gemm1<<<g1, 256, 0, stream>>>(x, Wct, XW);
    {
        size_t threads = (size_t)(EE + NN) * 32;
        k_agg<<<(unsigned)((threads + 255) / 256), 256, 0, stream>>>(ei, ew, deg,
                                                                     (const unsigned*)XW, AGG);
    }
    k_gemm2<<<NN / 16, 256, 0, stream>>>(AGG, Wlt, bl_i, bl_g, bl_o, out);
}
